// GraphAttention_82927228551580
// MI455X (gfx1250) — matrix operand order repaired and checked by reading, not compiled, not run on hardware
//
#include <hip/hip_runtime.h>

typedef __bf16 bf16;
typedef __attribute__((ext_vector_type(16))) bf16 bf16x16;
typedef __attribute__((ext_vector_type(8)))  bf16 bf16x8;
typedef __attribute__((ext_vector_type(8)))  float f32x8;

#define N_NODES 100000
#define B_TGT   1024
#define IN_DIM  384
#define LAT     128
#define FFN_DIM 512
#define SEGS    8
#define NTILES  64      // B_TGT/16
#define KCHUNKS 3125    // N_NODES/32

static __device__ __forceinline__ f32x8 zero8() {
    f32x8 r;
#pragma unroll
    for (int i = 0; i < 8; i++) r[i] = 0.f;
    return r;
}

static __device__ __forceinline__ bf16x16 cat8(bf16x8 lo, bf16x8 hi) {
    bf16x16 r;
#pragma unroll
    for (int i = 0; i < 8; i++) { r[i] = lo[i]; r[i + 8] = hi[i]; }
    return r;
}

static __device__ __forceinline__ f32x8 wmma_bf16(bf16x16 a, bf16x16 b, f32x8 c) {
    return __builtin_amdgcn_wmma_f32_16x16x32_bf16(false, a, false, b, (short)0, c, false, false);
}

// reductions across the 16-lane half of a wave32 (xor masks < 16 stay in-half)
static __device__ __forceinline__ float hmax16(float v) {
#pragma unroll
    for (int m = 1; m < 16; m <<= 1) v = fmaxf(v, __shfl_xor(v, m, 32));
    return v;
}
static __device__ __forceinline__ float hsum16(float v) {
#pragma unroll
    for (int m = 1; m < 16; m <<= 1) v += __shfl_xor(v, m, 32);
    return v;
}

// ---------------------------------------------------------------- kernel W
// one-time fp32 -> bf16 weight conversion (weights then stream as pure bf16)
__global__ __launch_bounds__(256) void k_cvtw(const float* __restrict__ src,
                                              bf16* __restrict__ dst, int n) {
    int i = blockIdx.x * 256 + threadIdx.x;
    if (i < n) dst[i] = (bf16)src[i];
}

// ---------------------------------------------------------------- kernel 0
// anyNbr[b] = any(neighbor_mask[b, :])
__global__ __launch_bounds__(256) void k_anynbr(const unsigned int* mask, int* anyNbr) {
    __shared__ unsigned int red[256];
    int b = blockIdx.x, t = threadIdx.x;
    const unsigned int* row = mask + (size_t)b * (N_NODES / 4);
    unsigned int v = 0;
    for (int i = t; i < N_NODES / 4; i += 256) v |= row[i];
    red[t] = v;
    __syncthreads();
    for (int s = 128; s > 0; s >>= 1) {
        if (t < s) red[t] |= red[t + s];
        __syncthreads();
    }
    if (t == 0) anyNbr[b] = (red[0] != 0);
}

// ---------------------------------------------------------------- kernel 1
// k_bf16[N][128] row-major, vT_bf16[128][N] transposed. kv tile staged in LDS.
__global__ __launch_bounds__(256) void k_kv(const float* __restrict__ x,
                                            const float* __restrict__ mem,
                                            const float* __restrict__ dtv,
                                            const bf16* __restrict__ Wkb, const float* __restrict__ bk,
                                            const bf16* __restrict__ Wvb, const float* __restrict__ bv,
                                            bf16* __restrict__ kbf, bf16* __restrict__ vT) {
    __shared__ bf16 kv[16 * IN_DIM];
    int n0 = blockIdx.x * 16, tid = threadIdx.x;
    for (int e = tid; e < 16 * IN_DIM; e += 256) {
        int r = e / IN_DIM, c = e - r * IN_DIM;
        int n = n0 + r;
        float f = (c < 128) ? x[(size_t)n * 128 + c]
                 : (c < 256) ? mem[(size_t)n * 128 + (c - 128)]
                             : dtv[(size_t)n * 128 + (c - 256)];
        kv[e] = (bf16)f;
    }
    __syncthreads();
    int w = tid >> 5, t = tid & 31, lc = t & 15, hf = t >> 4;
#pragma unroll
    for (int pass = 0; pass < 2; pass++) {
        int jb = w + pass * 8;
        bool isV = jb >= 8;
        int jt = isV ? jb - 8 : jb;
        int j = jt * 16 + lc;
        const bf16* W = isV ? Wvb : Wkb;
        const float* bias = isV ? bv : bk;
        f32x8 c = zero8();
#pragma unroll
        for (int kk = 0; kk < 12; kk++) {
            const bf16* ap = &kv[lc * IN_DIM + kk * 32 + hf * 8];
            bf16x16 a = cat8(*(const bf16x8*)ap, *(const bf16x8*)(ap + 16));
            const bf16* bp = &W[(size_t)j * IN_DIM + kk * 32 + hf * 8];
            bf16x16 b = cat8(*(const bf16x8*)bp, *(const bf16x8*)(bp + 16));
            c = wmma_bf16(a, b, c);
        }
        float bj = bias[j];
        if (!isV) {
#pragma unroll
            for (int r = 0; r < 8; r++)
                kbf[(size_t)(n0 + r + hf * 8) * LAT + j] = (bf16)(c[r] + bj);
        } else {
            bf16x8 pk;
#pragma unroll
            for (int r = 0; r < 8; r++) pk[r] = (bf16)(c[r] + bj);
            *(bf16x8*)&vT[(size_t)j * N_NODES + n0 + hf * 8] = pk;
        }
    }
}

// ---------------------------------------------------------------- kernel 2
// q_bf16[B][128] (pre-scaled by 1/sqrt(128)); also materializes q_input fp32.
__global__ __launch_bounds__(256) void k_q(const float* __restrict__ x,
                                           const float* __restrict__ mem,
                                           const float* __restrict__ ttv,
                                           const int* __restrict__ idx,
                                           const bf16* __restrict__ Wqb, const float* __restrict__ bq,
                                           bf16* __restrict__ qbf, float* __restrict__ qin) {
    __shared__ bf16 kv[16 * IN_DIM];
    __shared__ int rows[16];
    int b0 = blockIdx.x * 16, tid = threadIdx.x;
    if (tid < 16) rows[tid] = idx[b0 + tid];
    __syncthreads();
    for (int e = tid; e < 16 * IN_DIM; e += 256) {
        int r = e / IN_DIM, c = e - r * IN_DIM;
        int n = rows[r], b = b0 + r;
        float f = (c < 128) ? x[(size_t)n * 128 + c]
                 : (c < 256) ? mem[(size_t)n * 128 + (c - 128)]
                             : ttv[(size_t)b * 128 + (c - 256)];
        kv[e] = (bf16)f;
        qin[(size_t)b * IN_DIM + c] = f;
    }
    __syncthreads();
    int w = tid >> 5, t = tid & 31, lc = t & 15, hf = t >> 4;
    int j = w * 16 + lc;
    f32x8 c = zero8();
#pragma unroll
    for (int kk = 0; kk < 12; kk++) {
        const bf16* ap = &kv[lc * IN_DIM + kk * 32 + hf * 8];
        bf16x16 a = cat8(*(const bf16x8*)ap, *(const bf16x8*)(ap + 16));
        const bf16* bp = &Wqb[(size_t)j * IN_DIM + kk * 32 + hf * 8];
        bf16x16 b = cat8(*(const bf16x8*)bp, *(const bf16x8*)(bp + 16));
        c = wmma_bf16(a, b, c);
    }
    float bj = bq[j];
    const float scale = 0.08838834764831845f;  // 1/sqrt(128)
#pragma unroll
    for (int r = 0; r < 8; r++)
        qbf[(size_t)(b0 + r + hf * 8) * LAT + j] = (bf16)((c[r] + bj) * scale);
}

// ---------------------------------------------------------------- kernel 3
// mask-gated flash attention. grid(64 tiles, SEGS). 8 waves/block, each wave
// streams 32-key chunks with stride 64, skipping blocks with no mask/self bit.
__global__ __launch_bounds__(256) void k_attn(const bf16* __restrict__ qbf,
                                              const bf16* __restrict__ kbf,
                                              const bf16* __restrict__ vT,
                                              const unsigned char* __restrict__ mask,
                                              const int* __restrict__ anyNbr,
                                              const int* __restrict__ idx,
                                              float* __restrict__ pm, float* __restrict__ pl,
                                              float* __restrict__ pa) {
    __shared__ int selfLds[16];
    __shared__ unsigned char maskLds[8][16 * 32];
    __shared__ bf16 pLds[8][16 * 32];
    __shared__ float mSh[8][16], lSh[8][16];
    __shared__ float accBuf[16 * 128];
    int tileB = blockIdx.x, seg = blockIdx.y;
    int b0 = tileB * 16, tid = threadIdx.x;
    if (tid < 16) {
        int b = b0 + tid;
        selfLds[tid] = anyNbr[b] ? -1 : idx[b];
    }
    for (int e = tid; e < 16 * 128; e += 256) accBuf[e] = 0.f;
    __syncthreads();
    int w = tid >> 5, t = tid & 31, lc = t & 15, hf = t >> 4, off = hf * 8;

    bf16x16 qA[4];
#pragma unroll
    for (int kk = 0; kk < 4; kk++) {
        const bf16* p = &qbf[(size_t)(b0 + lc) * LAT + kk * 32 + off];
        qA[kk] = cat8(*(const bf16x8*)p, *(const bf16x8*)(p + 16));
    }
    int self8[8];
#pragma unroll
    for (int r = 0; r < 8; r++) self8[r] = selfLds[r + off];
    int svLane = selfLds[lc];

    const unsigned char* mlane = mask + (size_t)(b0 + lc) * N_NODES + hf * 16;

    float m[8], l[8];
    f32x8 acc[8];
#pragma unroll
    for (int r = 0; r < 8; r++) { m[r] = -1e30f; l[r] = 0.f; }
#pragma unroll
    for (int jt = 0; jt < 8; jt++) acc[jt] = zero8();

    for (int c = seg * 8 + w; c < KCHUNKS; c += SEGS * 8) {
        int n0 = c * 32;
        // prefetch this wave's next mask block (2KB ahead per row)
        if (c + SEGS * 8 < KCHUNKS)
            __builtin_prefetch(mlane + (size_t)(c + SEGS * 8) * 32, 0, 0);
        // probe 16x32 mask block: lane = (row lc, 16 cols at n0+hf*16)
        const unsigned char* mrow = mlane + n0;
        unsigned int w0 = *(const unsigned int*)(mrow + 0);
        unsigned int w1 = *(const unsigned int*)(mrow + 4);
        unsigned int w2 = *(const unsigned int*)(mrow + 8);
        unsigned int w3 = *(const unsigned int*)(mrow + 12);
        bool selfIn = (svLane >= n0) && (svLane < n0 + 32);
        if (!__any(((w0 | w1 | w2 | w3) != 0) || selfIn)) continue;

        // stash mask block into LDS (row-major) for column-wise reads
        unsigned char* md = &maskLds[w][lc * 32 + hf * 16];
        *(unsigned int*)(md + 0) = w0;
        *(unsigned int*)(md + 4) = w1;
        *(unsigned int*)(md + 8) = w2;
        *(unsigned int*)(md + 12) = w3;

        // S = q @ k^T for two 16x16 key tiles (K=128 -> 4 bf16 WMMAs each)
        f32x8 s0 = zero8(), s1 = zero8();
#pragma unroll
        for (int kk = 0; kk < 4; kk++) {
            const bf16* kp0 = &kbf[(size_t)(n0 + lc) * LAT + kk * 32 + hf * 8];
            const bf16* kp1 = &kbf[(size_t)(n0 + 16 + lc) * LAT + kk * 32 + hf * 8];
            s0 = wmma_bf16(qA[kk], cat8(*(const bf16x8*)kp0, *(const bf16x8*)(kp0 + 16)), s0);
            s1 = wmma_bf16(qA[kk], cat8(*(const bf16x8*)kp1, *(const bf16x8*)(kp1 + 16)), s1);
        }
        // mask + online softmax
        float rmax[8];
#pragma unroll
        for (int r = 0; r < 8; r++) {
            int row = r + off;
            unsigned char mb0 = maskLds[w][row * 32 + lc];
            unsigned char mb1 = maskLds[w][row * 32 + 16 + lc];
            int sr = self8[r];
            float a0 = (mb0 || (n0 + lc) == sr) ? s0[r] : -1e30f;
            float a1 = (mb1 || (n0 + 16 + lc) == sr) ? s1[r] : -1e30f;
            s0[r] = a0; s1[r] = a1;
            rmax[r] = fmaxf(a0, a1);
        }
        float sc[8];
#pragma unroll
        for (int r = 0; r < 8; r++) {
            float mn = fmaxf(m[r], hmax16(rmax[r]));
            sc[r] = __expf(m[r] - mn);
            m[r] = mn;
        }
        float psum[8];
#pragma unroll
        for (int r = 0; r < 8; r++) {
            float p0 = (s0[r] > -1e29f) ? __expf(s0[r] - m[r]) : 0.f;
            float p1 = (s1[r] > -1e29f) ? __expf(s1[r] - m[r]) : 0.f;
            pLds[w][(r + off) * 32 + lc] = (bf16)p0;
            pLds[w][(r + off) * 32 + 16 + lc] = (bf16)p1;
            psum[r] = p0 + p1;
        }
#pragma unroll
        for (int r = 0; r < 8; r++) l[r] = l[r] * sc[r] + hsum16(psum[r]);
#pragma unroll
        for (int jt = 0; jt < 8; jt++)
#pragma unroll
            for (int r = 0; r < 8; r++) acc[jt][r] *= sc[r];

        // P (column-per-lane) -> A-frag (row-per-lane) via LDS bounce
        const bf16* pp = &pLds[w][lc * 32 + off];
        bf16x16 pA = cat8(*(const bf16x8*)pp, *(const bf16x8*)(pp + 16));
        // acc += P @ V ; V read via transposed vT (contiguous per lane)
#pragma unroll
        for (int jt = 0; jt < 8; jt++) {
            const bf16* vp = &vT[(size_t)(jt * 16 + lc) * N_NODES + n0 + hf * 8];
            acc[jt] = wmma_bf16(pA, cat8(*(const bf16x8*)vp, *(const bf16x8*)(vp + 16)), acc[jt]);
        }
    }

    // intra-block merge of the 8 wave partials
    if (lc == 0) {
#pragma unroll
        for (int r = 0; r < 8; r++) { mSh[w][r + off] = m[r]; lSh[w][r + off] = l[r]; }
    }
    __syncthreads();
    float wgt[8];
#pragma unroll
    for (int r = 0; r < 8; r++) {
        float Mv = -1e30f;
        for (int ww = 0; ww < 8; ww++) Mv = fmaxf(Mv, mSh[ww][r + off]);
        wgt[r] = __expf(m[r] - Mv);
    }
#pragma unroll
    for (int jt = 0; jt < 8; jt++)
#pragma unroll
        for (int r = 0; r < 8; r++)
            atomicAdd(&accBuf[(r + off) * 128 + jt * 16 + lc], acc[jt][r] * wgt[r]);
    __syncthreads();
    size_t pbase = (size_t)tileB * SEGS + seg;
    if (tid < 16) {
        float Mv = -1e30f;
        for (int ww = 0; ww < 8; ww++) Mv = fmaxf(Mv, mSh[ww][tid]);
        float Lv = 0.f;
        for (int ww = 0; ww < 8; ww++) Lv += lSh[ww][tid] * __expf(mSh[ww][tid] - Mv);
        pm[pbase * 16 + tid] = Mv;
        pl[pbase * 16 + tid] = Lv;
    }
    for (int e = tid; e < 2048; e += 256) pa[pbase * 2048 + e] = accBuf[e];
}

// ---------------------------------------------------------------- kernel 4
// combine SEGS partials -> agg; z = relu([agg,q_input]@W1^T + b1)@W2^T + b2
__global__ __launch_bounds__(256) void k_ffn(const float* __restrict__ pm,
                                             const float* __restrict__ pl,
                                             const float* __restrict__ pa,
                                             const float* __restrict__ qin,
                                             const bf16* __restrict__ W1b, const float* __restrict__ b1,
                                             const bf16* __restrict__ W2b, const float* __restrict__ b2,
                                             float* __restrict__ out) {
    __shared__ bf16 hLds[16 * FFN_DIM];
    __shared__ bf16 h1Lds[16 * LAT];
    int tileB = blockIdx.x, b0 = tileB * 16, tid = threadIdx.x;
    for (int e = tid; e < 2048; e += 256) {
        int r = e >> 7;
        float Mv = -1e30f;
#pragma unroll
        for (int s = 0; s < SEGS; s++) Mv = fmaxf(Mv, pm[((size_t)tileB * SEGS + s) * 16 + r]);
        float Lv = 0.f, Vv = 0.f;
#pragma unroll
        for (int s = 0; s < SEGS; s++) {
            size_t pb = (size_t)tileB * SEGS + s;
            float wv = __expf(pm[pb * 16 + r] - Mv);
            Lv += pl[pb * 16 + r] * wv;
            Vv += pa[pb * 2048 + e] * wv;
        }
        hLds[r * FFN_DIM + (e & 127)] = (bf16)(Vv / Lv);
    }
    for (int e = tid; e < 16 * IN_DIM; e += 256) {
        int r = e / IN_DIM, c = e - r * IN_DIM;
        hLds[r * FFN_DIM + 128 + c] = (bf16)qin[(size_t)(b0 + r) * IN_DIM + c];
    }
    __syncthreads();
    int w = tid >> 5, t = tid & 31, lc = t & 15, hf = t >> 4, off = hf * 8;
    int j = w * 16 + lc;
    f32x8 c1 = zero8();
#pragma unroll
    for (int kk = 0; kk < 16; kk++) {
        const bf16* ap = &hLds[lc * FFN_DIM + kk * 32 + off];
        bf16x16 a = cat8(*(const bf16x8*)ap, *(const bf16x8*)(ap + 16));
        const bf16* bp = &W1b[(size_t)j * FFN_DIM + kk * 32 + hf * 8];
        bf16x16 b = cat8(*(const bf16x8*)bp, *(const bf16x8*)(bp + 16));
        c1 = wmma_bf16(a, b, c1);
    }
    float bj = b1[j];
#pragma unroll
    for (int r = 0; r < 8; r++) h1Lds[(r + off) * LAT + j] = (bf16)fmaxf(c1[r] + bj, 0.f);
    __syncthreads();
    f32x8 c2 = zero8();
#pragma unroll
    for (int kk = 0; kk < 4; kk++) {
        const bf16* ap = &h1Lds[lc * LAT + kk * 32 + off];
        bf16x16 a = cat8(*(const bf16x8*)ap, *(const bf16x8*)(ap + 16));
        const bf16* bp = &W2b[(size_t)j * LAT + kk * 32 + hf * 8];
        bf16x16 b = cat8(*(const bf16x8*)bp, *(const bf16x8*)(bp + 16));
        c2 = wmma_bf16(a, b, c2);
    }
    float b2j = b2[j];
#pragma unroll
    for (int r = 0; r < 8; r++) out[(size_t)(b0 + r + off) * LAT + j] = c2[r] + b2j;
}

// ---------------------------------------------------------------- launch
extern "C" void kernel_launch(void* const* d_in, const int* in_sizes, int n_in,
                              void* d_out, int out_size, void* d_ws, size_t ws_size,
                              hipStream_t stream) {
    (void)in_sizes; (void)n_in; (void)out_size; (void)ws_size;
    const float* x   = (const float*)d_in[0];
    const float* mem = (const float*)d_in[1];
    const float* dtv = (const float*)d_in[2];
    const float* ttv = (const float*)d_in[3];
    const unsigned char* mask = (const unsigned char*)d_in[4];
    const int* idx = (const int*)d_in[5];
    const float* Wq = (const float*)d_in[6];
    const float* bq = (const float*)d_in[7];
    const float* Wk = (const float*)d_in[8];
    const float* bk = (const float*)d_in[9];
    const float* Wv = (const float*)d_in[10];
    const float* bv = (const float*)d_in[11];
    const float* W1 = (const float*)d_in[12];
    const float* b1 = (const float*)d_in[13];
    const float* W2 = (const float*)d_in[14];
    const float* b2 = (const float*)d_in[15];

    // workspace layout (~57.8 MB total)
    char* ws = (char*)d_ws;
    size_t o = 0;
    bf16* kbf = (bf16*)(ws + o); o += (size_t)N_NODES * LAT * 2;
    bf16* vT  = (bf16*)(ws + o); o += (size_t)N_NODES * LAT * 2;
    bf16* qbf = (bf16*)(ws + o); o += (size_t)B_TGT * LAT * 2;
    float* qin = (float*)(ws + o); o += (size_t)B_TGT * IN_DIM * 4;
    int* anyN  = (int*)(ws + o);   o += (size_t)B_TGT * 4;
    float* pm  = (float*)(ws + o); o += (size_t)NTILES * SEGS * 16 * 4;
    float* pl  = (float*)(ws + o); o += (size_t)NTILES * SEGS * 16 * 4;
    float* pa  = (float*)(ws + o); o += (size_t)NTILES * SEGS * 2048 * 4;
    bf16* Wqb = (bf16*)(ws + o); o += (size_t)LAT * IN_DIM * 2;
    bf16* Wkb = (bf16*)(ws + o); o += (size_t)LAT * IN_DIM * 2;
    bf16* Wvb = (bf16*)(ws + o); o += (size_t)LAT * IN_DIM * 2;
    bf16* W1b = (bf16*)(ws + o); o += (size_t)LAT * FFN_DIM * 2;
    bf16* W2b = (bf16*)(ws + o); o += (size_t)LAT * LAT * 2;

    const int nw = LAT * IN_DIM;
    k_cvtw<<<(nw + 255) / 256, 256, 0, stream>>>(Wq, Wqb, nw);
    k_cvtw<<<(nw + 255) / 256, 256, 0, stream>>>(Wk, Wkb, nw);
    k_cvtw<<<(nw + 255) / 256, 256, 0, stream>>>(Wv, Wvb, nw);
    k_cvtw<<<(LAT * FFN_DIM + 255) / 256, 256, 0, stream>>>(W1, W1b, LAT * FFN_DIM);
    k_cvtw<<<(LAT * LAT + 255) / 256, 256, 0, stream>>>(W2, W2b, LAT * LAT);

    k_anynbr<<<B_TGT, 256, 0, stream>>>((const unsigned int*)mask, anyN);
    k_kv<<<N_NODES / 16, 256, 0, stream>>>(x, mem, dtv, Wkb, bk, Wvb, bv, kbf, vT);
    k_q<<<NTILES, 256, 0, stream>>>(x, mem, ttv, idx, Wqb, bq, qbf, qin);
    k_attn<<<dim3(NTILES, SEGS), 256, 0, stream>>>(qbf, kbf, vT, mask, anyN, idx, pm, pl, pa);
    k_ffn<<<NTILES, 256, 0, stream>>>(pm, pl, pa, qin, W1b, b1, W2b, b2, (float*)d_out);
}
